// MGMoEModel_3977139716882
// MI455X (gfx1250) — matrix operand order repaired and checked by reading, not compiled, not run on hardware
//
#include <hip/hip_runtime.h>
#include <hip/hip_bf16.h>

typedef __bf16 bf16;
typedef __attribute__((ext_vector_type(16))) __bf16 v16bf;
typedef __attribute__((ext_vector_type(8)))  __bf16 v8bf;
typedef __attribute__((ext_vector_type(16))) float  v16f;
typedef __attribute__((ext_vector_type(8)))  float  v8f;

#define BN_EPS 1e-5f

__device__ __forceinline__ float swishf(float x) { return x / (1.f + __expf(-x)); }

// gfx1250 async DMA: global memory -> LDS, 16 bytes per lane, tracked by ASYNCcnt.
__device__ __forceinline__ void async_ld_b128(void* lds_dst, const void* gsrc) {
  asm volatile("global_load_async_to_lds_b128 %0, %1, off"
               :: "v"((unsigned)(unsigned long long)lds_dst),   // LDS offset (addrspace(3) low bits)
                  "v"((unsigned long long)gsrc)                 // 64-bit global address
               : "memory");
}
__device__ __forceinline__ void wait_async0() {
  asm volatile("s_wait_asynccnt 0x0" ::: "memory");
}

// ---------------------------------------------------------------------------
// Transpose + f32->bf16:  src[b][K][N] (f32, row-major)  ->  dst[b][N][K] (bf16)
// grid = (N/32, K/32, batches), block = (32, 8)
// ---------------------------------------------------------------------------
__global__ __launch_bounds__(256) void k_transpose_cvt(const float* __restrict__ src,
                                                       bf16* __restrict__ dst,
                                                       int K, int N) {
  __shared__ float tile[32][33];
  const int n0 = blockIdx.x * 32, k0 = blockIdx.y * 32;
  const long long bo = (long long)blockIdx.z * K * N;
  const int tx = threadIdx.x, ty = threadIdx.y;
#pragma unroll
  for (int j = 0; j < 4; ++j)
    tile[ty + j * 8][tx] = src[bo + (long long)(k0 + ty + j * 8) * N + n0 + tx];
  __syncthreads();
#pragma unroll
  for (int j = 0; j < 4; ++j)
    dst[bo + (long long)(n0 + ty + j * 8) * K + k0 + tx] = (bf16)tile[tx][ty + j * 8];
}

// ---------------------------------------------------------------------------
// Fused eval-BatchNorm + swish, f32 -> bf16.
// ---------------------------------------------------------------------------
__global__ void k_bn_swish(const float* __restrict__ x, long long xmod,
                           const float* __restrict__ g, const float* __restrict__ b,
                           const float* __restrict__ m, const float* __restrict__ v,
                           int lastdim, long long batchElems,
                           bf16* __restrict__ out, long long total) {
  long long i = (long long)blockIdx.x * blockDim.x + threadIdx.x;
  if (i >= total) return;
  long long p = (i / batchElems) * lastdim + (i % lastdim);
  float y = (x[i % xmod] - m[p]) * (g[p] * rsqrtf(v[p] + BN_EPS)) + b[p];
  out[i] = (bf16)swishf(y);
}

// ---------------------------------------------------------------------------
// Batched bf16 WMMA GEMM:  C[z] = A[z] (MxK) x B[z] (KxN) (+ bias[z])
//   A: bf16 (async-DMA staged, double-buffered) or f32 (inline cvt when A_F32)
//   Bt: bf16 [N][K] row-major (B column-major) -> contiguous WMMA B-fragments
//   C: f32 row-major
// Block tile 128 x BN, BK=32, 8 waves (4x2), wave tile 32 x (BN/2).
// grid = (N/BN, M/128, batches), block = 256
// ---------------------------------------------------------------------------
template <int BN, bool A_F32, bool HAS_BIAS>
__global__ __launch_bounds__(256) void k_gemm_wmma(
    const void* __restrict__ Apv, long long sA, int lda,
    const bf16* __restrict__ Btp, long long sB, int ldb,
    float* __restrict__ Cp, long long sC, int ldc,
    const float* __restrict__ bias, int sBias, int K) {
  constexpr int NT = BN / 32;        // 16-wide B tiles per wave
  constexpr int DB = A_F32 ? 1 : 2;  // double-buffer LDS on the async path
  __shared__ alignas(64) bf16 As[DB * 128 * 32];
  __shared__ alignas(64) bf16 Bs[DB * BN * 32];

  const int tid  = threadIdx.x;
  const int lane = tid & 31;
  const int wv   = tid >> 5;
  const int wm   = wv & 3;   // 4 wave-rows of 32
  const int wn   = wv >> 2;  // 2 wave-cols of BN/2
  const int row0 = blockIdx.y * 128;
  const int col0 = blockIdx.x * BN;
  const long long z = blockIdx.z;

  const bf16* Bt = Btp + z * sB;
  float*      Cc = Cp  + z * sC;

  // WMMA fragment k order used here (contiguous; A and B match, not the ISA table order):
  //  A lane: row = lane&15, holds 16 contiguous K starting at kb=(lane>>4)*16
  //  B lane: col = lane&15, holds 16 contiguous K starting at (lane>>4)*16
  const int fr  = lane & 15;
  const int akb = (lane >> 4) * 16;
  const int bkb = (lane >> 4) * 16;

  v8f acc[2][NT] = {};

  auto mma_tile = [&](int cur) {
    const int aBase = cur * 128 * 32;
    const int bBase = cur * BN * 32;
    const int am0 = aBase + (wm * 32 + fr) * 32 + akb;
    const int am1 = aBase + (wm * 32 + 16 + fr) * 32 + akb;
    v8bf a0lo = *(const v8bf*)&As[am0];
    v8bf a0hi = *(const v8bf*)&As[am0 + 8];
    v8bf a1lo = *(const v8bf*)&As[am1];
    v8bf a1hi = *(const v8bf*)&As[am1 + 8];
    v16bf a0 = __builtin_shufflevector(a0lo, a0hi, 0,1,2,3,4,5,6,7,8,9,10,11,12,13,14,15);
    v16bf a1 = __builtin_shufflevector(a1lo, a1hi, 0,1,2,3,4,5,6,7,8,9,10,11,12,13,14,15);
#pragma unroll
    for (int ni = 0; ni < NT; ++ni) {
      v16bf b = *(const v16bf*)&Bs[bBase + (wn * (BN / 2) + ni * 16 + fr) * 32 + bkb];
      acc[0][ni] = __builtin_amdgcn_wmma_f32_16x16x32_bf16(false, a0, false, b, (short)0, acc[0][ni], false, false);
      acc[1][ni] = __builtin_amdgcn_wmma_f32_16x16x32_bf16(false, a1, false, b, (short)0, acc[1][ni], false, false);
    }
  };

  auto stageB = [&](int bi, int k0) {
#pragma unroll
    for (int c = 0; c < BN / 64; ++c) {  // 16B chunks: BN rows x 64B / 256 threads
      const int ch = tid + c * 256;
      const int r = ch >> 2, ko = (ch & 3) * 8;
      async_ld_b128(&Bs[bi * BN * 32 + r * 32 + ko],
                    Bt + (long long)(col0 + r) * ldb + k0 + ko);
    }
  };

  const int nk = K / 32;

  if constexpr (!A_F32) {
    const bf16* Aa = (const bf16*)Apv + z * sA + (long long)row0 * lda;
    auto stageA = [&](int bi, int k0) {
#pragma unroll
      for (int c = 0; c < 2; ++c) {      // 128 rows x 64B / 256 threads
        const int ch = tid + c * 256;
        const int r = ch >> 2, ko = (ch & 3) * 8;
        async_ld_b128(&As[bi * 128 * 32 + r * 32 + ko],
                      Aa + (long long)r * lda + k0 + ko);
      }
    };
    stageA(0, 0);
    stageB(0, 0);
    for (int kt = 0; kt < nk; ++kt) {
      const int cur = kt & 1;
      wait_async0();       // this wave's DMA into buf[cur] done
      __syncthreads();     // everyone's DMA done + previous reads of buf[cur^1] done
      if (kt + 1 < nk) {   // overlap next tile's DMA with this tile's WMMAs
        stageA(cur ^ 1, (kt + 1) * 32);
        stageB(cur ^ 1, (kt + 1) * 32);
      }
      mma_tile(cur);
    }
  } else {
    // f32 A (fc_0): load+convert through registers; B still via async DMA.
    const float* Aa = (const float*)Apv + z * sA;
    const int arow = tid >> 1, ako = (tid & 1) * 16;
    for (int kt = 0; kt < nk; ++kt) {
      const int k0 = kt * 32;
      v16f af = *(const v16f*)(Aa + (long long)(row0 + arow) * lda + k0 + ako);
      v16bf aval;
#pragma unroll
      for (int i = 0; i < 16; ++i) aval[i] = (bf16)af[i];
      __syncthreads();                 // previous tile fully consumed
      stageB(0, k0);
      *(v16bf*)&As[arow * 32 + ako] = aval;
      wait_async0();
      __syncthreads();
      mma_tile(0);
    }
  }

  // Epilogue. C/D layout: VGPR j -> M = j + (lane>>4)*8 ; N = lane&15
  const int mhi = (lane >> 4) * 8;
#pragma unroll
  for (int ni = 0; ni < NT; ++ni) {
    const int n = col0 + wn * (BN / 2) + ni * 16 + fr;
    float bv = 0.f;
    if constexpr (HAS_BIAS) bv = bias[z * sBias + wn * (BN / 2) + ni * 16 + fr];
#pragma unroll
    for (int j = 0; j < 8; ++j) {
      const long long mA = row0 + wm * 32 + j + mhi;
      Cc[mA * ldc + n]        = acc[0][ni][j] + bv;
      Cc[(mA + 16) * ldc + n] = acc[1][ni][j] + bv;
    }
  }
}

// ---------------------------------------------------------------------------
// Gates: logits[t,b,e] = h0[b,:] . gate_w[t,:,e] + gate_b[t,e]; softmax over E=8.
// ---------------------------------------------------------------------------
__global__ void k_gates(const float* __restrict__ h0, const float* __restrict__ gw,
                        const float* __restrict__ gb, float* __restrict__ gates,
                        int B, int F) {
  const int id = blockIdx.x * blockDim.x + threadIdx.x;  // t*B + b
  const int t = id / B, b = id % B;
  const float* hr  = h0 + (long long)b * F;
  const float* gwt = gw + (long long)t * F * 8;
  float acc[8] = {};
  for (int f = 0; f < F; ++f) {
    const float hv = hr[f];
#pragma unroll
    for (int e = 0; e < 8; ++e) acc[e] += hv * gwt[f * 8 + e];
  }
  float mx = -1e30f;
#pragma unroll
  for (int e = 0; e < 8; ++e) { acc[e] += gb[t * 8 + e]; mx = fmaxf(mx, acc[e]); }
  float s = 0.f;
#pragma unroll
  for (int e = 0; e < 8; ++e) { acc[e] = __expf(acc[e] - mx); s += acc[e]; }
  const float inv = 1.f / s;
#pragma unroll
  for (int e = 0; e < 8; ++e) gates[(long long)id * 8 + e] = acc[e] * inv;
}

// task_in[t,b,d] = sum_e gates[t,b,e] * eo[e,b,d]
__global__ void k_mix(const float* __restrict__ gates, const float* __restrict__ eo,
                      float* __restrict__ task_in, int B, int D, long long total) {
  long long i = (long long)blockIdx.x * blockDim.x + threadIdx.x;
  if (i >= total) return;
  const long long bd = (long long)B * D;
  const int t = (int)(i / bd);
  const long long r = i % bd;  // b*D + d
  const int b = (int)(r / D);
  const float* gp = gates + ((long long)t * B + b) * 8;
  float acc = 0.f;
#pragma unroll
  for (int e = 0; e < 8; ++e) acc += gp[e] * eo[(long long)e * bd + r];
  task_in[i] = acc;
}

// out[t,b,c] = sum_d swish(bn2(z[t,b,d])) * t_wf[t,d,c] + t_bf[t,c]   (NC=2)
__global__ void k_task_head(const float* __restrict__ z,
                            const float* __restrict__ g, const float* __restrict__ bb,
                            const float* __restrict__ m, const float* __restrict__ v,
                            const float* __restrict__ wf, const float* __restrict__ bf,
                            float* __restrict__ out, int B, int D) {
  const int id = blockIdx.x * blockDim.x + threadIdx.x;  // t*B + b
  const int t = id / B;
  const float* zr = z + (long long)id * D;
  const float* pg = g + t * D, *pb = bb + t * D, *pm = m + t * D, *pv = v + t * D;
  const float* wt = wf + (long long)t * D * 2;
  float a0 = 0.f, a1 = 0.f;
  for (int d = 0; d < D; ++d) {
    const float y = (zr[d] - pm[d]) * (pg[d] * rsqrtf(pv[d] + BN_EPS)) + pb[d];
    const float a = swishf(y);
    a0 += a * wt[d * 2 + 0];
    a1 += a * wt[d * 2 + 1];
  }
  out[(long long)id * 2 + 0] = a0 + bf[t * 2 + 0];
  out[(long long)id * 2 + 1] = a1 + bf[t * 2 + 1];
}

// ---------------------------------------------------------------------------
extern "C" void kernel_launch(void* const* d_in, const int* in_sizes, int n_in,
                              void* d_out, int out_size, void* d_ws, size_t ws_size,
                              hipStream_t stream) {
  (void)in_sizes; (void)n_in; (void)out_size; (void)ws_size;
  constexpr int B = 1024, C = 50, CS = 800, R = 64, F = 3200, D = 256, E = 8, T = 4;

  const float* x      = (const float*)d_in[0];
  const float* w0     = (const float*)d_in[1];
  const float* b0     = (const float*)d_in[2];
  const float* f_bn_g = (const float*)d_in[3];
  const float* f_bn_b = (const float*)d_in[4];
  const float* f_bn_m = (const float*)d_in[5];
  const float* f_bn_v = (const float*)d_in[6];
  const float* w1     = (const float*)d_in[7];
  const float* gate_w = (const float*)d_in[8];
  const float* gate_b = (const float*)d_in[9];
  const float* e_bn1_g = (const float*)d_in[10];
  const float* e_bn1_b = (const float*)d_in[11];
  const float* e_bn1_m = (const float*)d_in[12];
  const float* e_bn1_v = (const float*)d_in[13];
  const float* e_w1    = (const float*)d_in[14];
  const float* e_bn2_g = (const float*)d_in[15];
  const float* e_bn2_b = (const float*)d_in[16];
  const float* e_bn2_m = (const float*)d_in[17];
  const float* e_bn2_v = (const float*)d_in[18];
  const float* e_wf    = (const float*)d_in[19];
  const float* e_bf    = (const float*)d_in[20];
  const float* t_bn1_g = (const float*)d_in[21];
  const float* t_bn1_b = (const float*)d_in[22];
  const float* t_bn1_m = (const float*)d_in[23];
  const float* t_bn1_v = (const float*)d_in[24];
  const float* t_w1    = (const float*)d_in[25];
  const float* t_bn2_g = (const float*)d_in[26];
  const float* t_bn2_b = (const float*)d_in[27];
  const float* t_bn2_m = (const float*)d_in[28];
  const float* t_bn2_v = (const float*)d_in[29];
  const float* t_wf    = (const float*)d_in[30];
  const float* t_bf    = (const float*)d_in[31];
  float* out = (float*)d_out;

  // deterministic bump allocator over d_ws (~155 MB total)
  char* wp = (char*)d_ws;
  auto alloc = [&wp](size_t bytes) -> void* {
    void* r = (void*)wp;
    wp += (bytes + 255) & ~(size_t)255;
    return r;
  };
  bf16*  w0t     = (bf16*)alloc((size_t)C * R * CS * 2);
  bf16*  w1t     = (bf16*)alloc((size_t)F * F * 2);
  bf16*  ew1t    = (bf16*)alloc((size_t)E * D * F * 2);
  bf16*  ewft    = (bf16*)alloc((size_t)E * D * D * 2);
  bf16*  tw1t    = (bf16*)alloc((size_t)T * D * D * 2);
  float* h0      = (float*)alloc((size_t)B * F * 4);
  float* gates   = (float*)alloc((size_t)T * B * E * 4);
  bf16*  a1      = (bf16*)alloc((size_t)B * F * 2);
  float* h1      = (float*)alloc((size_t)B * F * 4);
  bf16*  ae      = (bf16*)alloc((size_t)E * B * F * 2);
  float* e1      = (float*)alloc((size_t)E * B * D * 4);
  bf16*  ae2     = (bf16*)alloc((size_t)E * B * D * 2);
  float* eo      = (float*)alloc((size_t)E * B * D * 4);
  float* task_in = (float*)alloc((size_t)T * B * D * 4);
  bf16*  at1     = (bf16*)alloc((size_t)T * B * D * 2);
  float* zbuf    = (float*)alloc((size_t)T * B * D * 4);

  const dim3 tb(32, 8);
  // weight transposes (f32 [K][N] -> bf16 [N][K])
  k_transpose_cvt<<<dim3(R / 32, CS / 32, C), tb, 0, stream>>>(w0, w0t, CS, R);
  k_transpose_cvt<<<dim3(F / 32, F / 32, 1), tb, 0, stream>>>(w1, w1t, F, F);
  k_transpose_cvt<<<dim3(D / 32, F / 32, E), tb, 0, stream>>>(e_w1, ew1t, F, D);
  k_transpose_cvt<<<dim3(D / 32, D / 32, E), tb, 0, stream>>>(e_wf, ewft, D, D);
  k_transpose_cvt<<<dim3(D / 32, D / 32, T), tb, 0, stream>>>(t_w1, tw1t, D, D);

  // fc_0: per-chunk GEMM, A = f32 x (inline cvt), bias b0
  k_gemm_wmma<64, true, true><<<dim3(R / 64, B / 128, C), 256, 0, stream>>>(
      (const void*)x, /*sA=*/CS, /*lda=*/C * CS,
      w0t, (long long)R * CS, CS,
      h0, /*sC=*/R, /*ldc=*/F,
      b0, /*sBias=*/R, /*K=*/CS);

  // gates (reads pre-fc_1 h0)
  k_gates<<<(T * B) / 256, 256, 0, stream>>>(h0, gate_w, gate_b, gates, B, F);

  // fc_1: a1 = swish(bn(h0)) -> bf16 ; h1 = a1 @ w1
  {
    long long tot = (long long)B * F;
    k_bn_swish<<<(unsigned)((tot + 255) / 256), 256, 0, stream>>>(
        h0, tot, f_bn_g, f_bn_b, f_bn_m, f_bn_v, F, tot, a1, tot);
  }
  k_gemm_wmma<128, false, false><<<dim3(F / 128, B / 128, 1), 256, 0, stream>>>(
      (const void*)a1, 0, F, w1t, 0, F, h1, 0, F, nullptr, 0, F);

  // experts L1: ae[e] = swish(bn1_e(h1)) ; e1[e] = ae[e] @ e_w1[e]
  {
    long long tot = (long long)E * B * F;
    k_bn_swish<<<(unsigned)((tot + 255) / 256), 256, 0, stream>>>(
        h1, (long long)B * F, e_bn1_g, e_bn1_b, e_bn1_m, e_bn1_v, F, (long long)B * F, ae, tot);
  }
  k_gemm_wmma<128, false, false><<<dim3(D / 128, B / 128, E), 256, 0, stream>>>(
      (const void*)ae, (long long)B * F, F,
      ew1t, (long long)D * F, F,
      e1, (long long)B * D, D, nullptr, 0, F);

  // experts L2: ae2 = swish(bn2_e(e1)) ; eo[e] = ae2[e] @ e_wf[e] + e_bf[e]
  {
    long long tot = (long long)E * B * D;
    k_bn_swish<<<(unsigned)((tot + 255) / 256), 256, 0, stream>>>(
        e1, tot, e_bn2_g, e_bn2_b, e_bn2_m, e_bn2_v, D, (long long)B * D, ae2, tot);
  }
  k_gemm_wmma<128, false, true><<<dim3(D / 128, B / 128, E), 256, 0, stream>>>(
      (const void*)ae2, (long long)B * D, D,
      ewft, (long long)D * D, D,
      eo, (long long)B * D, D, e_bf, D, D);

  // gate-weighted mix: task_in[t] = sum_e gates[t,:,e] * eo[e]
  {
    long long tot = (long long)T * B * D;
    k_mix<<<(unsigned)((tot + 255) / 256), 256, 0, stream>>>(gates, eo, task_in, B, D, tot);
  }

  // task L1: at1 = swish(bn1_t(task_in)) ; z[t] = at1[t] @ t_w1[t]
  {
    long long tot = (long long)T * B * D;
    k_bn_swish<<<(unsigned)((tot + 255) / 256), 256, 0, stream>>>(
        task_in, tot, t_bn1_g, t_bn1_b, t_bn1_m, t_bn1_v, D, (long long)B * D, at1, tot);
  }
  k_gemm_wmma<128, false, false><<<dim3(D / 128, B / 128, T), 256, 0, stream>>>(
      (const void*)at1, (long long)B * D, D,
      tw1t, (long long)D * D, D,
      zbuf, (long long)B * D, D, nullptr, 0, D);

  // task head: out[t,b,:2] = swish(bn2_t(z)) @ t_wf[t] + t_bf[t]
  k_task_head<<<(T * B) / 256, 256, 0, stream>>>(
      zbuf, t_bn2_g, t_bn2_b, t_bn2_m, t_bn2_v, t_wf, t_bf, out, B, D);
}
